// MultiHeadedCausalSelfAttentionBlock_81647328297556
// MI455X (gfx1250) — matrix operand order repaired and checked by reading, not compiled, not run on hardware
//
#include <hip/hip_runtime.h>
#include <hip/hip_bf16.h>

// ---------------------------------------------------------------------------
// Transformer block for MI455X (gfx1250, wave32, WMMA 16x16x32 f16, TDM).
// ---------------------------------------------------------------------------

typedef __attribute__((ext_vector_type(16))) _Float16 v16h;
typedef __attribute__((ext_vector_type(8)))  _Float16 v8h;
typedef __attribute__((ext_vector_type(8)))  float    v8f;
typedef __attribute__((ext_vector_type(4)))  unsigned int u32x4;
typedef __attribute__((ext_vector_type(8)))  int      i32x8;
typedef __attribute__((ext_vector_type(4)))  int      i32x4;

#define T_DIM   4096
#define E_DIM   2048
#define N_HEADS 16
#define HD_DIM  128
#define MLP_DIM 8192
#define LN_EPS  1e-5f

#if defined(__HIP_DEVICE_COMPILE__) && __has_builtin(__builtin_amdgcn_tensor_load_to_lds)
#define USE_TDM 1
#else
#define USE_TDM 0
#endif

__device__ __forceinline__ v16h mk16(v8h lo, v8h hi) {
    return __builtin_shufflevector(lo, hi, 0, 1, 2, 3, 4, 5, 6, 7,
                                           8, 9, 10, 11, 12, 13, 14, 15);
}

__device__ __forceinline__ v8f wmma_f16(v16h a, v16h b, v8f c) {
    // D = A(16x32 f16) * B(32x16 f16) + C(16x16 f32)
    return __builtin_amdgcn_wmma_f32_16x16x32_f16(
        /*neg_a=*/false, a, /*neg_b=*/false, b,
        /*c_mod=*/(short)0, c, /*reuse_a=*/false, /*reuse_b=*/false);
}

// ---------------------------------------------------------------------------
// Tensor Data Mover: DMA a 128-row x 32-col f16 tile (row stride K elements)
// from global memory into LDS.  2D descriptor: groups 2/3 zero-filled.
// This toolchain exposes the 6-arg builtin:
//   (u32x4 g0, i32x8 g1, i32x4, i32x4, i32x8, i32 cpol)
// ---------------------------------------------------------------------------
__device__ __forceinline__ void tdm_load_b_tile(const _Float16* gsrc,
                                                void* ldst, int K) {
#if USE_TDM
    const unsigned long long ga = (unsigned long long)(size_t)gsrc;
    const unsigned int ga_lo = (unsigned int)ga;
    const unsigned int ga_hi = (unsigned int)(ga >> 32);
    const unsigned int lds_addr = (unsigned int)(size_t)ldst;  // LDS offset bits
    const unsigned int td0 = (unsigned int)K;   // tensor_dim0 (elements)
    const unsigned int td1 = 1u << 20;          // tensor_dim1 (ample)
    u32x4 g0;
    g0[0] = 1u;                                   // count=1, user descriptor
    g0[1] = lds_addr;                             // lds_addr [63:32]
    g0[2] = ga_lo;                                // global_addr [95:64]
    g0[3] = (ga_hi & 0x01FFFFFFu) | (2u << 30);   // global_addr hi | type=2
    i32x8 g1;
    g1[0] = (int)(1u << 16);                                      // data_size=2B
    g1[1] = (int)((td0 & 0xFFFFu) << 16);                         // tdim0 lo
    g1[2] = (int)(((td0 >> 16) & 0xFFFFu) | ((td1 & 0xFFFFu) << 16));
    g1[3] = (int)(((td1 >> 16) & 0xFFFFu) | (32u << 16));         // tile_dim0=32
    g1[4] = (int)128u;                            // tile_dim1=128, tile_dim2=0
    g1[5] = (int)(unsigned int)K;                 // tensor_dim0_stride lo
    g1[6] = 0;                                    // stride0 hi | stride1 lo
    g1[7] = 0;                                    // stride1 hi
    i32x4 z4 = {};
    i32x8 z8 = {};
    __builtin_amdgcn_tensor_load_to_lds(g0, g1, z4, z4, z8, 0);
#else
    (void)gsrc; (void)ldst; (void)K;
#endif
}

__device__ __forceinline__ void tdm_wait() {
#if USE_TDM
    __builtin_amdgcn_s_wait_tensorcnt(0);
#endif
}

// fallback cooperative staging (256 threads copy 128x32 halfs)
__device__ __forceinline__ void coop_stage(const _Float16* src, _Float16* dst,
                                           int K, int tid) {
    const int row = tid >> 1, seg = (tid & 1) * 16;
    *(v16h*)(dst + row * 32 + seg) = *(const v16h*)(src + (size_t)row * K + seg);
}

// ---------------------------------------------------------------------------
// 1) global LayerNorm statistics (mean/var over all T*E elements)
// ---------------------------------------------------------------------------
__global__ void zero_stats_kernel(float* stats) {
    stats[0] = 0.f;
    stats[1] = 0.f;
}

__global__ __launch_bounds__(256) void reduce_stats_kernel(
    const float* __restrict__ x, float* __restrict__ stats, int n) {
    float s = 0.f, sq = 0.f;
    for (int i = blockIdx.x * blockDim.x + threadIdx.x; i < n;
         i += gridDim.x * blockDim.x) {
        float v = x[i];
        s += v;
        sq += v * v;
    }
    #pragma unroll
    for (int d = 1; d < 32; d <<= 1) {
        s  += __shfl_xor(s, d, 32);
        sq += __shfl_xor(sq, d, 32);
    }
    __shared__ float ss[8], sqs[8];
    const int wave = threadIdx.x >> 5, lane = threadIdx.x & 31;
    if (lane == 0) { ss[wave] = s; sqs[wave] = sq; }
    __syncthreads();
    if (threadIdx.x == 0) {
        float ts = 0.f, tq = 0.f;
        #pragma unroll
        for (int w = 0; w < 8; ++w) { ts += ss[w]; tq += sqs[w]; }
        atomicAdd(&stats[0], ts);
        atomicAdd(&stats[1], tq);
    }
}

__global__ __launch_bounds__(256) void ln_global_cast_kernel(
    const float* __restrict__ x, const float* __restrict__ sc,
    const float* __restrict__ lc, const float* __restrict__ stats,
    _Float16* __restrict__ g, int n) {
    const float inv_n = 1.0f / (float)((size_t)T_DIM * E_DIM);
    const float mean = stats[0] * inv_n;
    const float var  = stats[1] * inv_n - mean * mean;
    const float rstd = rsqrtf(var + LN_EPS);
    for (int i = blockIdx.x * blockDim.x + threadIdx.x; i < n;
         i += gridDim.x * blockDim.x) {
        const int c = i & (E_DIM - 1);
        g[i] = (_Float16)(((x[i] - mean) * rstd) * sc[c] + lc[c]);
    }
}

// ---------------------------------------------------------------------------
// 2) weight transpose + cast: wt[n*K + k] = (f16) w[k*N + n]
// ---------------------------------------------------------------------------
__global__ __launch_bounds__(256) void transpose_cast_kernel(
    const float* __restrict__ w, _Float16* __restrict__ wt, int K, int N) {
    __shared__ float tile[32][33];
    const int kb = blockIdx.y * 32, nb = blockIdx.x * 32;
    const int tx = threadIdx.x, ty = threadIdx.y;
    #pragma unroll
    for (int i = ty; i < 32; i += 8)
        tile[i][tx] = w[(size_t)(kb + i) * N + nb + tx];
    __syncthreads();
    #pragma unroll
    for (int i = ty; i < 32; i += 8)
        wt[(size_t)(nb + i) * K + kb + tx] = (_Float16)tile[tx][i];
}

// ---------------------------------------------------------------------------
// 3) WMMA GEMM:  C[M,N] = A[M,K] (f16 row major) * Bt[N,K] (f16, B transposed)
//    8 waves/block, wave tile 32x64 (2x4 WMMA tiles).  Per-block B tile
//    (128 x 32) is staged into LDS by the Tensor Data Mover, double-buffered
//    against WMMA compute (TENSORcnt + workgroup barrier).
// ---------------------------------------------------------------------------
template <bool BIAS, bool RELU, bool RES, bool OUTF16, bool OUTTRANS>
__global__ __launch_bounds__(256) void gemm_wmma_kernel(
    const _Float16* __restrict__ A, const _Float16* __restrict__ Bt,
    const float* __restrict__ bias, const float* __restrict__ res,
    float* __restrict__ outf, _Float16* __restrict__ outh,
    int M, int N, int K) {
    __shared__ alignas(32) _Float16 ldsB[2][128 * 32];  // 2 x 8 KiB
    const int lane = threadIdx.x & 31, wave = threadIdx.x >> 5;
    const int l16 = lane & 15, hf = lane >> 4;
    const int mb = blockIdx.y * 128 + (wave >> 1) * 32;
    const int nb_blk = blockIdx.x * 128;
    const int nbw = (wave & 1) * 64;  // wave's N offset inside block tile

    v8f z = {};
    v8f acc[2][4];
    #pragma unroll
    for (int i = 0; i < 2; ++i)
        #pragma unroll
        for (int j = 0; j < 4; ++j) acc[i][j] = z;

    const _Float16* ap0 = A + (size_t)(mb + l16) * K;
    const _Float16* ap1 = A + (size_t)(mb + 16 + l16) * K;
    const _Float16* bsrc = Bt + (size_t)nb_blk * K;

    // prologue: stage tile 0
#if USE_TDM
    if (wave == 0) {
        tdm_load_b_tile(bsrc, &ldsB[0][0], K);
        tdm_wait();
    }
#else
    coop_stage(bsrc, &ldsB[0][0], K, threadIdx.x);
#endif
    __syncthreads();

    for (int kc = 0; kc < K; kc += 32) {
        const int cur = (kc >> 5) & 1;
        const int nxt = cur ^ 1;
#if USE_TDM
        if (wave == 0 && kc + 32 < K)
            tdm_load_b_tile(bsrc + kc + 32, &ldsB[nxt][0], K);
#else
        if (kc + 32 < K) coop_stage(bsrc + kc + 32, &ldsB[nxt][0], K, threadIdx.x);
#endif
        __builtin_prefetch(ap0 + kc + 256, 0, 0);
        __builtin_prefetch(ap1 + kc + 256, 0, 0);
        v16h af[2];
        af[0] = mk16(*(const v8h*)(ap0 + kc + hf * 8),
                     *(const v8h*)(ap0 + kc + 16 + hf * 8));
        af[1] = mk16(*(const v8h*)(ap1 + kc + hf * 8),
                     *(const v8h*)(ap1 + kc + 16 + hf * 8));
        v16h bf[4];
        #pragma unroll
        for (int ni = 0; ni < 4; ++ni)
            bf[ni] = mk16(*(const v8h*)(&ldsB[cur][(nbw + ni * 16 + l16) * 32 + hf * 8]),
                          *(const v8h*)(&ldsB[cur][(nbw + ni * 16 + l16) * 32 + 16 + hf * 8]));
        #pragma unroll
        for (int mi = 0; mi < 2; ++mi)
            #pragma unroll
            for (int ni = 0; ni < 4; ++ni)
                acc[mi][ni] = wmma_f16(af[mi], bf[ni], acc[mi][ni]);
#if USE_TDM
        if (wave == 0) tdm_wait();
#endif
        __syncthreads();
    }

    #pragma unroll
    for (int mi = 0; mi < 2; ++mi) {
        #pragma unroll
        for (int ni = 0; ni < 4; ++ni) {
            const int col = nb_blk + nbw + ni * 16 + l16;
            #pragma unroll
            for (int r = 0; r < 8; ++r) {
                const int row = mb + mi * 16 + r + 8 * hf;
                float v = acc[mi][ni][r];
                if (BIAS) v += bias[col];
                if (RELU) v = fmaxf(v, 0.f);
                if (RES)  v += res[(size_t)row * N + col];
                if (OUTTRANS)     outh[(size_t)col * M + row] = (_Float16)v;
                else if (OUTF16)  outh[(size_t)row * N + col] = (_Float16)v;
                else              outf[(size_t)row * N + col] = v;
            }
        }
    }
}

// ---------------------------------------------------------------------------
// 4) flash attention (causal): one wave = 16 Q rows of one head.
//    qh/kh: [T, E] f16 row-major; vt: [E, T] f16 (V transposed per channel).
// ---------------------------------------------------------------------------
__global__ __launch_bounds__(256) void attn_kernel(
    const _Float16* __restrict__ qh, const _Float16* __restrict__ kh,
    const _Float16* __restrict__ vt, float* __restrict__ y) {
    __shared__ alignas(32) _Float16 pb[8][16][40];  // per-wave P tile (16x32)
    const int lane = threadIdx.x & 31, wave = threadIdx.x >> 5;
    const int l16 = lane & 15, hf = lane >> 4;
    const int head = blockIdx.x;
    const int q0 = blockIdx.y * 128 + wave * 16;

    const _Float16* qrow = qh + (size_t)(q0 + l16) * E_DIM + head * HD_DIM;
    v16h qa[4];
    #pragma unroll
    for (int c = 0; c < 4; ++c)
        qa[c] = *(const v16h*)(qrow + c * 32 + hf * 16);

    v8f z = {};
    v8f o[8];
    #pragma unroll
    for (int i = 0; i < 8; ++i) o[i] = z;
    float mrow[8], lrow[8], alpha[8];
    #pragma unroll
    for (int r = 0; r < 8; ++r) { mrow[r] = -1e30f; lrow[r] = 0.f; }

    const float rscale = 0.08838834764831845f;  // 1/sqrt(128)
    const int nsteps = (q0 + 15) / 32 + 1;

    for (int kb = 0; kb < nsteps; ++kb) {
        const int sb = kb * 32;
        // S = Q * K^T for 32 keys (2 N-tiles), contraction over 128 channels
        v8f s0 = z, s1 = z;
        #pragma unroll
        for (int c = 0; c < 4; ++c) {
            const _Float16* kp = kh + (size_t)(sb + l16) * E_DIM +
                                 head * HD_DIM + c * 32 + hf * 16;
            v16h b0 = *(const v16h*)kp;
            v16h b1 = *(const v16h*)(kp + (size_t)16 * E_DIM);
            s0 = wmma_f16(qa[c], b0, s0);
            s1 = wmma_f16(qa[c], b1, s1);
        }
        // online softmax; causal mask only on diagonal blocks (wave-uniform)
        if (sb + 31 > q0) {
            #pragma unroll
            for (int r = 0; r < 8; ++r) {
                const int qr = q0 + r + 8 * hf;
                float v0 = (sb + l16 <= qr)      ? s0[r] * rscale : -1e30f;
                float v1 = (sb + 16 + l16 <= qr) ? s1[r] * rscale : -1e30f;
                float mx = fmaxf(v0, v1);
                #pragma unroll
                for (int d = 1; d < 16; d <<= 1)
                    mx = fmaxf(mx, __shfl_xor(mx, d, 32));
                const float nm = fmaxf(mrow[r], mx);
                const float al = __expf(mrow[r] - nm);
                const float p0 = __expf(v0 - nm);
                const float p1 = __expf(v1 - nm);
                float rs = p0 + p1;
                #pragma unroll
                for (int d = 1; d < 16; d <<= 1) rs += __shfl_xor(rs, d, 32);
                lrow[r] = lrow[r] * al + rs;
                mrow[r] = nm;
                alpha[r] = al;
                pb[wave][r + 8 * hf][l16]      = (_Float16)p0;
                pb[wave][r + 8 * hf][l16 + 16] = (_Float16)p1;
            }
        } else {
            #pragma unroll
            for (int r = 0; r < 8; ++r) {
                const float v0 = s0[r] * rscale;
                const float v1 = s1[r] * rscale;
                float mx = fmaxf(v0, v1);
                #pragma unroll
                for (int d = 1; d < 16; d <<= 1)
                    mx = fmaxf(mx, __shfl_xor(mx, d, 32));
                const float nm = fmaxf(mrow[r], mx);
                const float al = __expf(mrow[r] - nm);
                const float p0 = __expf(v0 - nm);
                const float p1 = __expf(v1 - nm);
                float rs = p0 + p1;
                #pragma unroll
                for (int d = 1; d < 16; d <<= 1) rs += __shfl_xor(rs, d, 32);
                lrow[r] = lrow[r] * al + rs;
                mrow[r] = nm;
                alpha[r] = al;
                pb[wave][r + 8 * hf][l16]      = (_Float16)p0;
                pb[wave][r + 8 * hf][l16 + 16] = (_Float16)p1;
            }
        }
        // rescale running O by alpha (per row)
        #pragma unroll
        for (int nt = 0; nt < 8; ++nt)
            #pragma unroll
            for (int r = 0; r < 8; ++r) o[nt][r] *= alpha[r];
        // O += P(16x32) * V(32x128)
        v16h pa = mk16(*(const v8h*)(&pb[wave][l16][hf * 8]),
                       *(const v8h*)(&pb[wave][l16][16 + hf * 8]));
        #pragma unroll
        for (int nt = 0; nt < 8; ++nt) {
            v16h vb = mk16(*(const v8h*)(vt + (size_t)(head * HD_DIM + nt * 16 + l16) * T_DIM +
                                         sb + hf * 8),
                           *(const v8h*)(vt + (size_t)(head * HD_DIM + nt * 16 + l16) * T_DIM +
                                         sb + 16 + hf * 8));
            o[nt] = wmma_f16(pa, vb, o[nt]);
        }
    }

    float inv[8];
    #pragma unroll
    for (int r = 0; r < 8; ++r) inv[r] = 1.0f / lrow[r];
    #pragma unroll
    for (int nt = 0; nt < 8; ++nt)
        #pragma unroll
        for (int r = 0; r < 8; ++r)
            y[(size_t)(q0 + r + 8 * hf) * E_DIM + head * HD_DIM + nt * 16 + l16] =
                o[nt][r] * inv[r];
}

// ---------------------------------------------------------------------------
// 5) x1 = x + y, per-row LayerNorm -> n (f16); x1 stored to d_out for residual
// ---------------------------------------------------------------------------
__global__ __launch_bounds__(256) void add_rowln_kernel(
    const float* __restrict__ x, const float* __restrict__ y,
    const float* __restrict__ sc, const float* __restrict__ lc,
    float* __restrict__ x1, _Float16* __restrict__ nh) {
    const int t = blockIdx.x;
    float vals[8];
    float s = 0.f, sq = 0.f;
    #pragma unroll
    for (int i = 0; i < 8; ++i) {
        const int e = threadIdx.x + i * 256;
        const float v = x[(size_t)t * E_DIM + e] + y[(size_t)t * E_DIM + e];
        vals[i] = v;
        s += v;
        sq += v * v;
    }
    #pragma unroll
    for (int d = 1; d < 32; d <<= 1) {
        s  += __shfl_xor(s, d, 32);
        sq += __shfl_xor(sq, d, 32);
    }
    __shared__ float ss[8], sqs[8];
    const int wave = threadIdx.x >> 5, lane = threadIdx.x & 31;
    if (lane == 0) { ss[wave] = s; sqs[wave] = sq; }
    __syncthreads();
    float ts = 0.f, tq = 0.f;
    #pragma unroll
    for (int w = 0; w < 8; ++w) { ts += ss[w]; tq += sqs[w]; }
    const float mu = ts * (1.0f / E_DIM);
    const float rstd = rsqrtf(tq * (1.0f / E_DIM) - mu * mu + LN_EPS);
    #pragma unroll
    for (int i = 0; i < 8; ++i) {
        const int e = threadIdx.x + i * 256;
        x1[(size_t)t * E_DIM + e] = vals[i];
        nh[(size_t)t * E_DIM + e] =
            (_Float16)((vals[i] - mu) * rstd * sc[e] + lc[e]);
    }
}

// ---------------------------------------------------------------------------
// workspace layout
// ---------------------------------------------------------------------------
static constexpr size_t SZ_TE_H = (size_t)T_DIM * E_DIM * 2;    // 16 MiB
static constexpr size_t SZ_TE_F = (size_t)T_DIM * E_DIM * 4;    // 32 MiB
static constexpr size_t SZ_EE_H = (size_t)E_DIM * E_DIM * 2;    //  8 MiB
static constexpr size_t SZ_EM_H = (size_t)E_DIM * MLP_DIM * 2;  // 32 MiB

static constexpr size_t OFF_STATS = 0;
static constexpr size_t OFF_G   = 256;                   // g, later reused for n
static constexpr size_t OFF_Q   = OFF_G + SZ_TE_H;
static constexpr size_t OFF_K   = OFF_Q + SZ_TE_H;
static constexpr size_t OFF_VT  = OFF_K + SZ_TE_H;
static constexpr size_t OFF_Y   = OFF_VT + SZ_TE_H;
static constexpr size_t OFF_QMT = OFF_Y + SZ_TE_F;
static constexpr size_t OFF_KMT = OFF_QMT + SZ_EE_H;
static constexpr size_t OFF_VMT = OFF_KMT + SZ_EE_H;
static constexpr size_t OFF_W1T = OFF_VMT + SZ_EE_H;
static constexpr size_t OFF_W2T = OFF_W1T + SZ_EM_H;
static constexpr size_t OFF_H   = OFF_Q;  // 64 MiB hidden aliases dead Q/K/VT/Y

extern "C" void kernel_launch(void* const* d_in, const int* in_sizes, int n_in,
                              void* d_out, int out_size, void* d_ws,
                              size_t ws_size, hipStream_t stream) {
    (void)in_sizes; (void)n_in; (void)out_size; (void)ws_size;
    const float* x       = (const float*)d_in[0];
    const float* qm      = (const float*)d_in[1];
    const float* km      = (const float*)d_in[2];
    const float* vm      = (const float*)d_in[3];
    const float* w1      = (const float*)d_in[4];
    const float* b1      = (const float*)d_in[5];
    const float* w2      = (const float*)d_in[6];
    const float* b2      = (const float*)d_in[7];
    const float* ln1_loc = (const float*)d_in[8];
    const float* ln1_sc  = (const float*)d_in[9];
    const float* ln2_loc = (const float*)d_in[10];
    const float* ln2_sc  = (const float*)d_in[11];
    float* out = (float*)d_out;

    char* ws = (char*)d_ws;
    float*     stats = (float*)(ws + OFF_STATS);
    _Float16*  g     = (_Float16*)(ws + OFF_G);
    _Float16*  qh    = (_Float16*)(ws + OFF_Q);
    _Float16*  kh    = (_Float16*)(ws + OFF_K);
    _Float16*  vth   = (_Float16*)(ws + OFF_VT);
    float*     yb    = (float*)(ws + OFF_Y);
    _Float16*  qmt   = (_Float16*)(ws + OFF_QMT);
    _Float16*  kmt   = (_Float16*)(ws + OFF_KMT);
    _Float16*  vmt   = (_Float16*)(ws + OFF_VMT);
    _Float16*  w1t   = (_Float16*)(ws + OFF_W1T);
    _Float16*  w2t   = (_Float16*)(ws + OFF_W2T);
    _Float16*  hbuf  = (_Float16*)(ws + OFF_H);
    _Float16*  nh    = g;  // reuse g region for pre-MLP LN output

    const int nelem = T_DIM * E_DIM;

    // global layernorm + cast
    zero_stats_kernel<<<1, 1, 0, stream>>>(stats);
    reduce_stats_kernel<<<1024, 256, 0, stream>>>(x, stats, nelem);
    ln_global_cast_kernel<<<2048, 256, 0, stream>>>(x, ln1_sc, ln1_loc, stats,
                                                    g, nelem);

    // weight transpose+cast to f16 (Wt[N][K])
    dim3 tb(32, 8);
    transpose_cast_kernel<<<dim3(E_DIM / 32, E_DIM / 32), tb, 0, stream>>>(qm, qmt, E_DIM, E_DIM);
    transpose_cast_kernel<<<dim3(E_DIM / 32, E_DIM / 32), tb, 0, stream>>>(km, kmt, E_DIM, E_DIM);
    transpose_cast_kernel<<<dim3(E_DIM / 32, E_DIM / 32), tb, 0, stream>>>(vm, vmt, E_DIM, E_DIM);
    transpose_cast_kernel<<<dim3(MLP_DIM / 32, E_DIM / 32), tb, 0, stream>>>(w1, w1t, E_DIM, MLP_DIM);
    transpose_cast_kernel<<<dim3(E_DIM / 32, MLP_DIM / 32), tb, 0, stream>>>(w2, w2t, MLP_DIM, E_DIM);

    // QKV projections (V stored transposed for attention PV fragments)
    dim3 ggrid(E_DIM / 128, T_DIM / 128);
    gemm_wmma_kernel<false, false, false, true, false><<<ggrid, 256, 0, stream>>>(
        g, qmt, nullptr, nullptr, nullptr, qh, T_DIM, E_DIM, E_DIM);
    gemm_wmma_kernel<false, false, false, true, false><<<ggrid, 256, 0, stream>>>(
        g, kmt, nullptr, nullptr, nullptr, kh, T_DIM, E_DIM, E_DIM);
    gemm_wmma_kernel<false, false, false, false, true><<<ggrid, 256, 0, stream>>>(
        g, vmt, nullptr, nullptr, nullptr, vth, T_DIM, E_DIM, E_DIM);

    // causal flash attention
    attn_kernel<<<dim3(N_HEADS, T_DIM / 128), 256, 0, stream>>>(qh, kh, vth, yb);

    // residual + per-row layernorm (x1 -> d_out, n -> f16)
    add_rowln_kernel<<<T_DIM, 256, 0, stream>>>(x, yb, ln2_sc, ln2_loc, out, nh);

    // MLP
    gemm_wmma_kernel<true, true, false, true, false>
        <<<dim3(MLP_DIM / 128, T_DIM / 128), 256, 0, stream>>>(
            nh, w1t, b1, nullptr, nullptr, hbuf, T_DIM, MLP_DIM, E_DIM);
    gemm_wmma_kernel<true, false, true, false, false>
        <<<dim3(E_DIM / 128, T_DIM / 128), 256, 0, stream>>>(
            hbuf, w2t, b2, out, out, nullptr, T_DIM, E_DIM, MLP_DIM);
}
